// QuantumEncoder_65481071399061
// MI455X (gfx1250) — matrix operand order repaired and checked by reading, not compiled, not run on hardware
//
#include <hip/hip_runtime.h>

typedef __attribute__((ext_vector_type(16))) _Float16 v16h;
typedef __attribute__((ext_vector_type(8)))  float    v8f;

#define N_QUBITS 20
#define LATENT   1024
#define REPS     3

// Composition of the 3 reverse-linear CX entangling layers, applied to an
// output index in reverse gate order: amp_final[j] = amp0[cx_perm(j)].
__device__ __forceinline__ uint32_t cx_perm(uint32_t j) {
  uint32_t x = j;
  #pragma unroll
  for (int r = 0; r < REPS; ++r) {
    #pragma unroll
    for (int i = 0; i < N_QUBITS - 1; ++i) {
      x ^= ((x >> i) & 1u) << (i + 1);   // CX(control=i, target=i+1)
    }
  }
  return x;   // stays < 2^20
}

// One wave computes one 16(j) x 16(b) output tile via
//   log|out| = base[b] + Bits(16x32,{0,1}) @ Diff(32x16)
// using two v_wmma_f32_16x16x32_f16 (hi + f16-residual of Diff) chained on
// the same f32 accumulator for ~f32 precision.
__global__ __launch_bounds__(256)
void qenc_wmma_kernel(const float* __restrict__ theta, float* __restrict__ out) {
  const int lane  = threadIdx.x & 31;
  const int wid   = (int)((blockIdx.x * blockDim.x + threadIdx.x) >> 5);
  const int jtile = wid >> 2;     // 0..63  (1024 / 16)
  const int btile = wid & 3;      // 0..3   (64 / 16)
  const int lo16  = lane & 15;
  const int hi    = lane >> 4;    // half-wave selector per WMMA layout

  // ---- B operand: Diff[k][b] = log|sin(th/2)| - log|cos(th/2)|, split f16 ----
  // B (32x16 f16), this code's contiguous k order: lanes 0-15 hold K=0..15 at N=lane; lanes 16-31 K=16..31.
  const int    b  = btile * 16 + lo16;
  const float* th = theta + b * N_QUBITS;

  float base = 0.0f;                       // sum_q log|cos(th_q/2)|
  #pragma unroll
  for (int q = 0; q < N_QUBITS; ++q) {
    float h = 0.5f * th[q];
    base += logf(fmaxf(fabsf(cosf(h)), 1e-38f));
  }

  v16h bhi, blo;
  #pragma unroll
  for (int e = 0; e < 16; ++e) {
    int   q = hi * 16 + e;                 // K index held by this lane/element
    float d = 0.0f;
    if (q < N_QUBITS) {
      float h  = 0.5f * th[q];
      float lc = logf(fmaxf(fabsf(cosf(h)), 1e-38f));
      float ls = logf(fmaxf(fabsf(sinf(h)), 1e-38f));
      d = ls - lc;
    }
    _Float16 dh = (_Float16)d;
    bhi[e] = dh;
    blo[e] = (_Float16)(d - (float)dh);    // residual: recovers f32 accuracy
  }

  // ---- A operand: {0,1} bits of the permuted basis index (exact in f16) ----
  // A (16x32 f16), same contiguous k order as B: lane L<16: M=L, K=0..15; L>=16: M=L-16, K=16..31.
  const uint32_t x = cx_perm((uint32_t)(jtile * 16 + lo16));
  v16h a;
  #pragma unroll
  for (int e = 0; e < 16; ++e) {
    int q = hi * 16 + e;
    a[e] = (_Float16)((x >> q) & 1u);      // bits >=20 are 0 -> K-padding
  }

  // ---- two chained WMMAs: acc = A@Bhi + A@Blo ----
  v8f acc = {};
  acc = __builtin_amdgcn_wmma_f32_16x16x32_f16(false, a, false, bhi,
                                               (short)0, acc, false, false);
  acc = __builtin_amdgcn_wmma_f32_16x16x32_f16(false, a, false, blo,
                                               (short)0, acc, false, false);

  // ---- epilogue: out[b][j] = exp(acc + base)  (always <= 1, no overflow) ----
  // D layout: VGPR v -> M = v + 8*hi (j index), N = lo16 (b index) == this lane's b.
  const int j0 = jtile * 16 + hi * 8;
  float*    op = out + b * LATENT + j0;    // 32B-aligned: j0 multiple of 8

  float4 r0, r1;
  r0.x = expf(acc[0] + base);
  r0.y = expf(acc[1] + base);
  r0.z = expf(acc[2] + base);
  r0.w = expf(acc[3] + base);
  r1.x = expf(acc[4] + base);
  r1.y = expf(acc[5] + base);
  r1.z = expf(acc[6] + base);
  r1.w = expf(acc[7] + base);
  *(float4*)(op)     = r0;
  *(float4*)(op + 4) = r1;
}

extern "C" void kernel_launch(void* const* d_in, const int* in_sizes, int n_in,
                              void* d_out, int out_size, void* d_ws, size_t ws_size,
                              hipStream_t stream) {
  const float* theta = (const float*)d_in[0];   // (64, 20) f32
  float*       out   = (float*)d_out;           // (64, 1024) f32
  // 256 output tiles of 16x16 -> 256 waves -> 32 blocks x 256 threads (8 waves).
  qenc_wmma_kernel<<<dim3(32), dim3(256), 0, stream>>>(theta, out);
}
